// PDTSP_Decoder_50079318671393
// MI455X (gfx1250) — matrix operand order repaired and checked by reading, not compiled, not run on hardware
//
#include <hip/hip_runtime.h>
#include <hip/hip_bf16.h>
#include <math.h>

// Problem constants (match reference)
#define B_    32
#define R_    200
#define N_    200
#define D_    128
#define H_    8
#define QK_   16
#define KNN_  10
#define CLIP_ 10.0f
#define M_    (B_ * R_)   // 6400 == B_*N_ as well

typedef __bf16 bf16;
typedef __attribute__((ext_vector_type(16))) __bf16 v16bf;
typedef __attribute__((ext_vector_type(8)))  float  v8f;

static __device__ __forceinline__ v8f wmma_bf16(const v16bf& a, const v16bf& b, const v8f& c) {
  return __builtin_amdgcn_wmma_f32_16x16x32_bf16(
      /*neg_a=*/false, a, /*neg_b=*/false, b,
      /*c_mod=*/(short)0, c, /*reuse_a=*/false, /*reuse_b=*/false);
}

// ---------------------------------------------------------------------------
// f32 -> bf16 conversion (row-major copy)
// ---------------------------------------------------------------------------
__global__ void cvt_bf16_kernel(const float* __restrict__ src,
                                bf16* __restrict__ dst, int n) {
  int i = blockIdx.x * blockDim.x + threadIdx.x;
  if (i < n) dst[i] = (bf16)src[i];
}

// f32 [K,N] row-major -> bf16 [N,K] row-major (transpose), so WMMA B-operand
// lanes read contiguous K-pairs.
__global__ void cvt_bf16_T_kernel(const float* __restrict__ src,
                                  bf16* __restrict__ dst, int K, int N) {
  int i = blockIdx.x * blockDim.x + threadIdx.x;
  if (i < K * N) {
    int k = i / N, n = i - k * N;
    dst[(size_t)n * K + k] = (bf16)src[i];
  }
}

// ---------------------------------------------------------------------------
// Top-K (K=10 smallest masked distances) + gather + pad-avg.
// One block (256 threads) per (b,r). Writes:
//   A1  [B*R, KNN*D]  bf16 : MLP input rows
//   Qin [B*R, 2*D]    bf16 : left half = current-node embedding
// ---------------------------------------------------------------------------
__global__ __launch_bounds__(256) void topk_gather_kernel(
    const float* __restrict__ enc,   // [B,N,D]
    const float* __restrict__ dist,  // [B,N,N]
    const float* __restrict__ mask,  // [B,R,N]
    const int*   __restrict__ cur,   // [B,R]
    bf16* __restrict__ A1,
    bf16* __restrict__ Qin) {
  const int row = blockIdx.x;          // b*R + r
  const int b   = row / R_;
  const int t   = threadIdx.x;

  __shared__ float dsh[256];
  __shared__ float rv[256];
  __shared__ int   ri[256];
  __shared__ int   knn_s[KNN_];
  __shared__ float mean_s[D_];
  __shared__ int   cnt_s;

  const int c = cur[row];
  const float* mrow = mask + (size_t)row * N_;

  float dv = INFINITY;
  if (t < N_) {
    float m = mrow[t];
    float d = dist[((size_t)b * N_ + c) * N_ + t];
    dv = isfinite(m) ? d : INFINITY;   // visited -> +inf
  }
  dsh[t] = dv;
  __syncthreads();

  // K iterative argmins; tie-break on smaller index (matches lax.top_k).
  for (int k = 0; k < KNN_; k++) {
    rv[t] = dsh[t]; ri[t] = t;
    __syncthreads();
    for (int s = 128; s > 0; s >>= 1) {
      if (t < s) {
        float ov = rv[t + s]; int oi = ri[t + s];
        if (ov < rv[t] || (ov == rv[t] && oi < ri[t])) { rv[t] = ov; ri[t] = oi; }
      }
      __syncthreads();
    }
    if (t == 0) {
      int bi = ri[0];
      knn_s[k] = (rv[0] == INFINITY) ? N_ : bi;   // pad index N for masked slots
      dsh[bi] = INFINITY;
    }
    __syncthreads();
  }

  if (t == 0) {
    int cn = 0;
    for (int k = 0; k < KNN_; k++) if (knn_s[k] < N_) cn++;
    cnt_s = cn;
  }
  __syncthreads();

  // mean of valid gathered rows (pad-avg replacement value)
  if (t < D_) {
    float s = 0.f;
    for (int k = 0; k < KNN_; k++) {
      int idx = knn_s[k];
      if (idx < N_) s += enc[((size_t)b * N_ + idx) * D_ + t];
    }
    mean_s[t] = s / fmaxf((float)cnt_s, 1e-9f);
  }
  __syncthreads();

  bf16* arow = A1 + (size_t)row * (KNN_ * D_);
  for (int i = t; i < KNN_ * D_; i += 256) {
    int k = i >> 7, d = i & (D_ - 1);
    int idx = knn_s[k];
    float v = (idx < N_) ? enc[((size_t)b * N_ + idx) * D_ + d] : mean_s[d];
    arow[i] = (bf16)v;
  }
  if (t < D_) {
    Qin[(size_t)row * (2 * D_) + t] = (bf16)enc[((size_t)b * N_ + c) * D_ + t];
  }
}

// ---------------------------------------------------------------------------
// Generic bf16 WMMA GEMM:  C[M,Ncols] = act(A[M,K] @ W[K,Ncols] + bias)
//   A  : bf16 row-major, leading dim lda
//   Wt : bf16 *transposed* weights [Ncols, K], leading dim ldw (= K)
//   C  : bf16 row-major, leading dim ldc
// 256 threads = 8 waves; each wave owns one 32x32 output tile (2x2 register
// blocking: 2 A-frags x 2 B-frags -> 4 WMMAs per 32-wide K-step).
// Fragment layouts follow CDNA5 ISA 7.12.2 (16-bit A 16x32 / B 32x16).
// M, Ncols multiples of 32; K multiple of 32 (true for all call sites).
// ---------------------------------------------------------------------------
__global__ __launch_bounds__(256) void gemm_bf16_wmma_kernel(
    const bf16* __restrict__ A, int lda,
    const bf16* __restrict__ Wt, int ldw,
    const float* __restrict__ bias,
    bf16* __restrict__ C, int ldc,
    int M, int Ncols, int Ktot, int relu) {
  const int lane = threadIdx.x & 31;
  const int wave = threadIdx.x >> 5;
  const int ntn  = Ncols >> 5;           // 32-wide N tiles
  const int ntiles = (M >> 5) * ntn;
  const int tile = blockIdx.x * 8 + wave;
  if (tile >= ntiles) return;
  const int tm = tile / ntn;
  const int tn = tile - tm * ntn;

  const int l15    = lane & 15;
  const int kgrpA  = (lane >> 4) * 8;    // A: lanes 0-15 hold K 0-7/16-23; 16-31 hold 8-15/24-31
  const int khalfB = (lane >> 4) * 8;    // B: same K layout as A (0-7/16-23; 8-15/24-31)

  const bf16* arow0 = A  + (size_t)(tm * 32 + l15) * lda;
  const bf16* arow1 = arow0 + (size_t)16 * lda;
  const bf16* wrow0 = Wt + (size_t)(tn * 32 + l15) * ldw;
  const bf16* wrow1 = wrow0 + (size_t)16 * ldw;

  v8f acc00 = {}, acc01 = {}, acc10 = {}, acc11 = {};
  for (int k0 = 0; k0 < Ktot; k0 += 32) {
    __builtin_prefetch(arow0 + k0 + 64, 0, 1);   // next-next A tiles
    __builtin_prefetch(arow1 + k0 + 64, 0, 1);
    v16bf a0, a1, b0, b1;
#pragma unroll
    for (int j = 0; j < 4; j++) {
      int klo = k0 + kgrpA + 2 * j;
      int khi = k0 + 16 + kgrpA + 2 * j;
      a0[2 * j]         = arow0[klo];
      a0[2 * j + 1]     = arow0[klo + 1];
      a0[8 + 2 * j]     = arow0[khi];
      a0[8 + 2 * j + 1] = arow0[khi + 1];
      a1[2 * j]         = arow1[klo];
      a1[2 * j + 1]     = arow1[klo + 1];
      a1[8 + 2 * j]     = arow1[khi];
      a1[8 + 2 * j + 1] = arow1[khi + 1];
    }
#pragma unroll
    for (int j = 0; j < 8; j++) {
      int kk = k0 + khalfB + (j & 3) * 2 + (j >> 2) * 16;
      b0[2 * j]     = wrow0[kk];
      b0[2 * j + 1] = wrow0[kk + 1];
      b1[2 * j]     = wrow1[kk];
      b1[2 * j + 1] = wrow1[kk + 1];
    }
    acc00 = wmma_bf16(a0, b0, acc00);
    acc01 = wmma_bf16(a0, b1, acc01);
    acc10 = wmma_bf16(a1, b0, acc10);
    acc11 = wmma_bf16(a1, b1, acc11);
  }

  // Epilogue: C/D layout -> element i: row = base + i + (lane>=16 ? 8:0),
  // col = base + (lane&15).
  const int c0  = tn * 32 + l15;
  const int c1  = c0 + 16;
  const float bv0 = bias ? bias[c0] : 0.0f;
  const float bv1 = bias ? bias[c1] : 0.0f;
  const int r0 = tm * 32 + ((lane >> 4) ? 8 : 0);
  const int r1 = r0 + 16;
#pragma unroll
  for (int i = 0; i < 8; i++) {
    float v00 = acc00[i] + bv0, v01 = acc01[i] + bv1;
    float v10 = acc10[i] + bv0, v11 = acc11[i] + bv1;
    if (relu) {
      v00 = fmaxf(v00, 0.0f); v01 = fmaxf(v01, 0.0f);
      v10 = fmaxf(v10, 0.0f); v11 = fmaxf(v11, 0.0f);
    }
    C[(size_t)(r0 + i) * ldc + c0] = (bf16)v00;
    C[(size_t)(r0 + i) * ldc + c1] = (bf16)v01;
    C[(size_t)(r1 + i) * ldc + c0] = (bf16)v10;
    C[(size_t)(r1 + i) * ldc + c1] = (bf16)v11;
  }
}

// ---------------------------------------------------------------------------
// Fused attention + output projection + final score/softmax.
// One block (256 threads) per (b,r). Writes out[b,r,:] (f32).
// ---------------------------------------------------------------------------
__global__ __launch_bounds__(256) void attn_final_kernel(
    const bf16*  __restrict__ Qb,    // [B*R,128]
    const bf16*  __restrict__ Kb,    // [B*N,128]
    const bf16*  __restrict__ Vb,    // [B*N,128]
    const float* __restrict__ Wmhc,  // [128,128]
    const float* __restrict__ bmhc,  // [128]
    const float* __restrict__ enc,   // [B,N,D] f32
    const float* __restrict__ mask,  // [B,R,N]
    float* __restrict__ out) {       // [B,R,N]
  const int row = blockIdx.x;        // b*R + r
  const int b   = row / R_;
  const int t   = threadIdx.x;

  __shared__ float qs[D_];
  __shared__ float sc[H_ * N_];
  __shared__ float hred[H_];
  __shared__ float os[D_];
  __shared__ float mhs[D_];
  __shared__ float ls[N_];
  __shared__ float red[256];

  if (t < D_) qs[t] = (float)Qb[(size_t)row * D_ + t];
  __syncthreads();

  const float* mrow = mask + (size_t)row * N_;

  // scores: [H, N] = q . k / sqrt(QK) + mask
  if (t < N_) {
    const bf16* krow = Kb + ((size_t)b * N_ + t) * D_;
    float m = mrow[t];
#pragma unroll
    for (int h = 0; h < H_; h++) {
      float s = 0.f;
#pragma unroll
      for (int j = 0; j < QK_; j++) s += qs[h * QK_ + j] * (float)krow[h * QK_ + j];
      sc[h * N_ + t] = s * 0.25f + m;    // 1/sqrt(16) = 0.25
    }
  }
  __syncthreads();

  // per-head softmax over N
  if (t < H_) {
    float mx = -INFINITY;
    for (int n = 0; n < N_; n++) mx = fmaxf(mx, sc[t * N_ + n]);
    hred[t] = mx;
  }
  __syncthreads();
  for (int i = t; i < H_ * N_; i += 256) {
    int h = i / N_;
    sc[i] = __expf(sc[i] - hred[h]);
  }
  __syncthreads();
  if (t < H_) {
    float s = 0.f;
    for (int n = 0; n < N_; n++) s += sc[t * N_ + n];
    hred[t] = s;
  }
  __syncthreads();
  for (int i = t; i < H_ * N_; i += 256) {
    int h = i / N_;
    sc[i] /= hred[h];
  }
  __syncthreads();

  // out[h*16+d] = sum_n w[h,n] * v[n,h,d]
  if (t < D_) {
    int h = t >> 4, dd = t & 15;
    float acc = 0.f;
    for (int n = 0; n < N_; n++)
      acc += sc[h * N_ + n] * (float)Vb[((size_t)b * N_ + n) * D_ + h * QK_ + dd];
    os[t] = acc;
  }
  __syncthreads();

  // mh = out @ Wmhc + b_mhc  (f32 weights)
  if (t < D_) {
    float acc = bmhc[t];
    for (int j = 0; j < D_; j++) acc += os[j] * Wmhc[j * D_ + t];
    mhs[t] = acc;
  }
  __syncthreads();

  // logits[n] = CLIP*tanh(mh . enc[b,n] / sqrt(D)) + mask[n]
  if (t < N_) {
    const float* erow = enc + ((size_t)b * N_ + t) * D_;
    float s = 0.f;
    for (int d = 0; d < D_; d++) s += mhs[d] * erow[d];
    s *= 0.0883883476483184f;            // 1/sqrt(128)
    ls[t] = CLIP_ * tanhf(s) + mrow[t];
  }
  __syncthreads();

  // final softmax over N (tree reduction over 256 padded slots)
  red[t] = (t < N_) ? ls[t] : -INFINITY;
  __syncthreads();
  for (int s = 128; s > 0; s >>= 1) {
    if (t < s) red[t] = fmaxf(red[t], red[t + s]);
    __syncthreads();
  }
  const float mx = red[0];
  __syncthreads();
  const float ev = (t < N_) ? __expf(ls[t] - mx) : 0.f;
  red[t] = ev;
  __syncthreads();
  for (int s = 128; s > 0; s >>= 1) {
    if (t < s) red[t] += red[t + s];
    __syncthreads();
  }
  const float denom = red[0];
  if (t < N_) out[(size_t)row * N_ + t] = ev / denom;
}

// ---------------------------------------------------------------------------
// Host launcher
// ---------------------------------------------------------------------------
extern "C" void kernel_launch(void* const* d_in, const int* in_sizes, int n_in,
                              void* d_out, int out_size, void* d_ws, size_t ws_size,
                              hipStream_t stream) {
  (void)in_sizes; (void)n_in; (void)out_size; (void)ws_size;
  const float* enc  = (const float*)d_in[0];
  const float* dist = (const float*)d_in[1];
  const float* mask = (const float*)d_in[2];
  const int*   cur  = (const int*)  d_in[3];
  const float* Wq   = (const float*)d_in[4];
  const float* Wk   = (const float*)d_in[5];
  const float* Wv   = (const float*)d_in[6];
  const float* Wmhc = (const float*)d_in[7];
  const float* bmhc = (const float*)d_in[8];
  const float* W1   = (const float*)d_in[9];
  const float* b1   = (const float*)d_in[10];
  const float* W2   = (const float*)d_in[11];
  const float* b2   = (const float*)d_in[12];
  float* out = (float*)d_out;

  // workspace carve-up (256B aligned)
  char* ws = (char*)d_ws;
  size_t off = 0;
  auto carve = [&](size_t bytes) -> char* {
    char* p = ws + off;
    off += (bytes + 255) & ~(size_t)255;
    return p;
  };
  bf16* A1  = (bf16*)carve((size_t)M_ * KNN_ * D_ * 2);  // 16.4 MB
  bf16* Hb  = (bf16*)carve((size_t)M_ * 5 * D_ * 2);     //  8.2 MB
  bf16* Qin = (bf16*)carve((size_t)M_ * 2 * D_ * 2);     //  3.3 MB
  bf16* Qb  = (bf16*)carve((size_t)M_ * D_ * 2);
  bf16* Kb  = (bf16*)carve((size_t)M_ * D_ * 2);
  bf16* Vb  = (bf16*)carve((size_t)M_ * D_ * 2);
  bf16* Eb  = (bf16*)carve((size_t)B_ * N_ * D_ * 2);
  bf16* W1t = (bf16*)carve((size_t)KNN_ * D_ * 5 * D_ * 2);
  bf16* W2t = (bf16*)carve((size_t)5 * D_ * D_ * 2);
  bf16* Wqt = (bf16*)carve((size_t)2 * D_ * D_ * 2);
  bf16* Wkt = (bf16*)carve((size_t)D_ * D_ * 2);
  bf16* Wvt = (bf16*)carve((size_t)D_ * D_ * 2);

  const int TB = 256;
  // --- conversions ---
  {
    int n = B_ * N_ * D_;
    cvt_bf16_kernel<<<(n + TB - 1) / TB, TB, 0, stream>>>(enc, Eb, n);
  }
  cvt_bf16_T_kernel<<<(KNN_ * D_ * 5 * D_ + TB - 1) / TB, TB, 0, stream>>>(W1, W1t, KNN_ * D_, 5 * D_);
  cvt_bf16_T_kernel<<<(5 * D_ * D_ + TB - 1) / TB, TB, 0, stream>>>(W2, W2t, 5 * D_, D_);
  cvt_bf16_T_kernel<<<(2 * D_ * D_ + TB - 1) / TB, TB, 0, stream>>>(Wq, Wqt, 2 * D_, D_);
  cvt_bf16_T_kernel<<<(D_ * D_ + TB - 1) / TB, TB, 0, stream>>>(Wk, Wkt, D_, D_);
  cvt_bf16_T_kernel<<<(D_ * D_ + TB - 1) / TB, TB, 0, stream>>>(Wv, Wvt, D_, D_);

  // --- top-K + gather + pad-avg + cur_emb ---
  topk_gather_kernel<<<M_, TB, 0, stream>>>(enc, dist, mask, cur, A1, Qin);

  // --- GEMM chain (all bf16 WMMA, f32 accumulate, 32x32 tiles per wave) ---
  auto gemm = [&](const bf16* A, int lda, const bf16* Wt, int ldw,
                  const float* bias, bf16* C, int ldc,
                  int M, int Ncols, int K, int relu) {
    int tiles = (M >> 5) * (Ncols >> 5);
    gemm_bf16_wmma_kernel<<<(tiles + 7) / 8, TB, 0, stream>>>(
        A, lda, Wt, ldw, bias, C, ldc, M, Ncols, K, relu);
  };
  // h = relu(A1 @ W1 + b1)            [6400,1280]x[1280,640]
  gemm(A1, KNN_ * D_, W1t, KNN_ * D_, b1, Hb, 5 * D_, M_, 5 * D_, KNN_ * D_, 1);
  // unv = h @ W2 + b2 -> right half of qin (ldc = 2D)
  gemm(Hb, 5 * D_, W2t, 5 * D_, b2, Qin + D_, 2 * D_, M_, D_, 5 * D_, 0);
  // q = qin @ Wq                       [6400,256]x[256,128]
  gemm(Qin, 2 * D_, Wqt, 2 * D_, nullptr, Qb, D_, M_, D_, 2 * D_, 0);
  // k = enc @ Wk, v = enc @ Wv         [6400,128]x[128,128]
  gemm(Eb, D_, Wkt, D_, nullptr, Kb, D_, M_, D_, D_, 0);
  gemm(Eb, D_, Wvt, D_, nullptr, Vb, D_, M_, D_, D_, 0);

  // --- fused attention + final softmax ---
  attn_final_kernel<<<M_, TB, 0, stream>>>(Qb, Kb, Vb, Wmhc, bmhc, enc, mask, out);
}
